// Decoder_31044023616500
// MI455X (gfx1250) — matrix operand order repaired and checked by reading, not compiled, not run on hardware
//
#include <hip/hip_runtime.h>
#include <hip/hip_bf16.h>

// ---------------------------------------------------------------------------
// MI455X (gfx1250, wave32) implementation.
//  - WMMA bf16 (v_wmma_f32_16x16x32_bf16) for all GEMMs, f32 accumulate.
//  - Persistent 2-WG producer/consumer pipeline for the sequential LSTM scan.
//  - Attention-score kernel stages P/Q tiles into LDS with the Tensor Data
//    Mover (tensor_load_to_lds + s_wait_tensorcnt), using TDM's LDS-padding
//    feature to produce a bank-conflict-free 68-float row stride.
// ---------------------------------------------------------------------------

typedef __attribute__((ext_vector_type(16))) __bf16 v16bf;
typedef __attribute__((ext_vector_type(8)))  float  v8f;

#if __has_builtin(__builtin_amdgcn_tensor_load_to_lds) && \
    __has_builtin(__builtin_amdgcn_s_wait_tensorcnt)
#define USE_TDM 1
typedef __attribute__((ext_vector_type(4))) unsigned int tdm_g0_t;
typedef __attribute__((ext_vector_type(8))) int          tdm_g1_t;
typedef __attribute__((ext_vector_type(4))) int          tdm_g2_t;
#endif

#define DEVINL __device__ __forceinline__

DEVINL unsigned short f2bf(float f) {
  unsigned u = __float_as_uint(f);
  u += 0x7FFFu + ((u >> 16) & 1u);          // round-to-nearest-even
  return (unsigned short)(u >> 16);
}
DEVINL float bf2f(unsigned short h) { return __uint_as_float(((unsigned)h) << 16); }

DEVINL float fast_sig(float x)  { return __builtin_amdgcn_rcpf(1.f + __expf(-x)); }
DEVINL float fast_tanh(float x) { return 1.f - 2.f * __builtin_amdgcn_rcpf(__expf(2.f * x) + 1.f); }

DEVINL float dot8_bf(uint4 p, const float* v) {
  float s;
  s  = bf2f((unsigned short)(p.x & 0xffffu)) * v[0];
  s += bf2f((unsigned short)(p.x >> 16))     * v[1];
  s += bf2f((unsigned short)(p.y & 0xffffu)) * v[2];
  s += bf2f((unsigned short)(p.y >> 16))     * v[3];
  s += bf2f((unsigned short)(p.z & 0xffffu)) * v[4];
  s += bf2f((unsigned short)(p.z >> 16))     * v[5];
  s += bf2f((unsigned short)(p.w & 0xffffu)) * v[6];
  s += bf2f((unsigned short)(p.w >> 16))     * v[7];
  return s;
}

#ifdef USE_TDM
// LDS byte offset of a __shared__ object (generic -> AS3 -> integer).
DEVINL unsigned lds_off(const void* p) {
  return (unsigned)(unsigned long long)(__attribute__((address_space(3))) const void*)p;
}
#endif

// ---------------------------------------------------------------------------
// Elementwise conversion / prep kernels
// ---------------------------------------------------------------------------
__global__ void cvt_bf16_kernel(unsigned short* __restrict__ dst,
                                const float* __restrict__ src, int n) {
  for (int i = blockIdx.x * blockDim.x + threadIdx.x; i < n; i += gridDim.x * blockDim.x)
    dst[i] = f2bf(src[i]);
}

__global__ void cvt_pad_kernel(unsigned short* __restrict__ dst,
                               const float* __restrict__ src,
                               int rows_real, int rows_pad, int cols) {
  int n = rows_pad * cols;
  for (int i = blockIdx.x * blockDim.x + threadIdx.x; i < n; i += gridDim.x * blockDim.x) {
    int r = i / cols;
    dst[i] = (r < rows_real) ? f2bf(src[i]) : (unsigned short)0;
  }
}

__global__ void prep_kernel(float* __restrict__ b0, const float* __restrict__ bih0,
                            const float* __restrict__ bhh0,
                            float* __restrict__ b1, const float* __restrict__ bih1,
                            const float* __restrict__ bhh1, int n, int* __restrict__ counter) {
  int i = blockIdx.x * blockDim.x + threadIdx.x;
  if (i < n) { b0[i] = bih0[i] + bhh0[i]; b1[i] = bih1[i] + bhh1[i]; }
  if (i == 0) *counter = 0;   // reset pipeline counter every launch (graph-safe)
}

// ---------------------------------------------------------------------------
// Generic WMMA GEMM:  C[M,N] = act(A[M,K] * B[N,K]^T + bias + Cadd)
// A, B bf16 row-major; one 16x16 tile per wave; K multiple of 32.
// ---------------------------------------------------------------------------
__global__ void __launch_bounds__(256) wmma_gemm_kernel(
    const unsigned short* __restrict__ A, int lda,
    const unsigned short* __restrict__ B, int ldb,
    float* __restrict__ Cf, unsigned short* __restrict__ Cb, int ldc,
    const float* __restrict__ bias, const float* __restrict__ Cadd,
    int M, int Npad, int Nreal, int K, int relu) {
  const int wave = (blockIdx.x * blockDim.x + threadIdx.x) >> 5;
  const int lane = threadIdx.x & 31;
  const int tilesN = Npad >> 4;
  const int tilesM = M >> 4;
  if (wave >= tilesM * tilesN) return;          // wave-uniform: EXEC stays all-1s
  const int tm = wave / tilesN;
  const int tn = wave % tilesN;
  const int l  = lane & 15;
  const int hi = lane >> 4;

  const unsigned short* arow = A + (size_t)(tm * 16 + l) * lda;
  const unsigned short* brow = B + (size_t)(tn * 16 + l) * ldb;

  v8f acc = {};
  union { v16bf v; uint4 q[2]; } fa, fb;

  for (int k0 = 0; k0 < K; k0 += 32) {
    // A 16x32 bf16 fragment: lane row m = l, K = {hi*8..+7} U {16+hi*8..+7}
    fa.q[0] = *(const uint4*)(arow + k0 + hi * 8);
    fa.q[1] = *(const uint4*)(arow + k0 + 16 + hi * 8);
    // B 32x16 fragment: lane col n = l, K = {hi*8..+7} U {16+hi*8..+7}
    fb.q[0] = *(const uint4*)(brow + k0 + hi * 8);
    fb.q[1] = *(const uint4*)(brow + k0 + 16 + hi * 8);
    if (k0 + 32 < K) {
      __builtin_prefetch(arow + k0 + 32, 0, 1);   // global_prefetch for next K tile
      __builtin_prefetch(brow + k0 + 32, 0, 1);
    }
    acc = __builtin_amdgcn_wmma_f32_16x16x32_bf16(
        /*neg_a=*/false, fa.v, /*neg_b=*/false, fb.v,
        /*c_mod=*/(short)0, acc, /*reuse_a=*/false, /*reuse_b=*/false);
  }

  const int n = tn * 16 + l;
  if (n >= Nreal) return;
  const float bv = bias ? bias[n] : 0.f;
#pragma unroll
  for (int r = 0; r < 8; ++r) {
    const int m = tm * 16 + hi * 8 + r;           // C layout: m = 8*hi + r, n = l
    float v = acc[r] + bv;
    if (Cadd) v += Cadd[(size_t)m * ldc + n];
    if (relu) v = fmaxf(v, 0.f);
    if (Cf) Cf[(size_t)m * ldc + n] = v;
    else    Cb[(size_t)m * ldc + n] = f2bf(v);
  }
}

// ---------------------------------------------------------------------------
// Pipelined 2-layer LSTM scan. Block 0 = layer 0 (input proj precomputed as
// xg0 = x@Wih0^T + b0 via WMMA), block 1 = layer 1, synchronized through a
// release/acquire step counter in global memory.
// ---------------------------------------------------------------------------
__global__ void __launch_bounds__(1024, 1) lstm_kernel(
    const float* __restrict__ xg0,            // [T,1024] precomputed gates input
    const unsigned short* __restrict__ whh0,  // [1024,256] bf16
    const unsigned short* __restrict__ wih1,  // [1024,256] bf16
    const unsigned short* __restrict__ whh1,  // [1024,256] bf16
    const float* __restrict__ b1,             // [1024]
    float* __restrict__ h1seq,                // [T,256] layer-0 output stream
    unsigned short* __restrict__ dec_bf,      // [T,256] layer-1 output (bf16)
    int* __restrict__ counter, int T) {
  __shared__ float g_s[1024];
  __shared__ float h_s[256];
  __shared__ float c_s[256];
  __shared__ float x_s[256];
  const int j = threadIdx.x;
  if (j < 256) { h_s[j] = 0.f; c_s[j] = 0.f; }
  __syncthreads();

  if (blockIdx.x == 0) {
    // -------- layer 0 producer --------
    const unsigned short* w = whh0 + (size_t)j * 256;
    for (int t = 0; t < T; ++t) {
      float acc = xg0[(size_t)t * 1024 + j];
#pragma unroll 4
      for (int k = 0; k < 256; k += 8)
        acc += dot8_bf(*(const uint4*)(w + k), &h_s[k]);
      g_s[j] = acc;
      __syncthreads();
      if (j < 256) {
        const float iv = fast_sig (g_s[j]);
        const float fv = fast_sig (g_s[256 + j]);
        const float gv = fast_tanh(g_s[512 + j]);
        const float ov = fast_sig (g_s[768 + j]);
        const float c  = fv * c_s[j] + iv * gv;
        const float h  = ov * fast_tanh(c);
        c_s[j] = c; h_s[j] = h;
        h1seq[(size_t)t * 256 + j] = h;
        __threadfence();
      }
      __syncthreads();
      if (j == 0)
        __hip_atomic_store(counter, t + 1, __ATOMIC_RELEASE, __HIP_MEMORY_SCOPE_AGENT);
    }
  } else {
    // -------- layer 1 consumer --------
    const unsigned short* wa = wih1 + (size_t)j * 256;
    const unsigned short* wb = whh1 + (size_t)j * 256;
    for (int t = 0; t < T; ++t) {
      if (j == 0) {
        while (__hip_atomic_load(counter, __ATOMIC_ACQUIRE, __HIP_MEMORY_SCOPE_AGENT) < t + 1)
          __builtin_amdgcn_s_sleep(1);
      }
      __syncthreads();
      if (j < 256) x_s[j] = h1seq[(size_t)t * 256 + j];
      __syncthreads();
      float acc = b1[j];
#pragma unroll 4
      for (int k = 0; k < 256; k += 8) {
        acc += dot8_bf(*(const uint4*)(wa + k), &x_s[k]);
        acc += dot8_bf(*(const uint4*)(wb + k), &h_s[k]);
      }
      g_s[j] = acc;
      __syncthreads();
      if (j < 256) {
        const float iv = fast_sig (g_s[j]);
        const float fv = fast_sig (g_s[256 + j]);
        const float gv = fast_tanh(g_s[512 + j]);
        const float ov = fast_sig (g_s[768 + j]);
        const float c  = fv * c_s[j] + iv * gv;
        const float h  = ov * fast_tanh(c);
        c_s[j] = c; h_s[j] = h;
        dec_bf[(size_t)t * 256 + j] = f2bf(h);
      }
      __syncthreads();
    }
  }
}

// ---------------------------------------------------------------------------
// scores[a,b] = sum_l tanh(Pfull[a,l] + Q[b,l]) * wy2[l] + by2
// where Pfull = dec@Wy1[:, :W]^T + x@Wy1[:, W:3W]^T + by1 (fused in GEMM).
// 16x16 output tile per block; 16x64 f32 tiles staged into LDS by the TDM
// (2-D tile descriptor, pad 4 DWORDs every 64 -> 68-float row stride).
// ---------------------------------------------------------------------------
__global__ void __launch_bounds__(256) scores_kernel(
    const float* __restrict__ pfull, const float* __restrict__ qm,
    const float* __restrict__ wy2,   const float* __restrict__ by2,
    float* __restrict__ scores, int Nn, int Ll) {
  __shared__ float Pt[16][68];   // 68*4B row stride: 16B-aligned rows, no bank conflicts
  __shared__ float Qt[16][68];
  __shared__ float Wv[64];
  const int tx = threadIdx.x & 15;    // b within tile
  const int ty = threadIdx.x >> 4;    // a within tile
  const int a0 = blockIdx.y * 16;
  const int b0 = blockIdx.x * 16;
#ifdef USE_TDM
  const unsigned ldsP = lds_off(&Pt[0][0]);
  const unsigned ldsQ = lds_off(&Qt[0][0]);
#endif
  float s = 0.f;
  for (int l0 = 0; l0 < Ll; l0 += 64) {
#ifdef USE_TDM
    if (threadIdx.x == 0) {
      // D# group1: data_size=4B (2<<16), pad_enable (1<<20),
      // pad_interval=64 DWORDs (5<<22), pad_amount=4 DWORDs (3<<25),
      // tensor_dim0=Ll, tensor_dim1=Ll, tile_dim0=64, tile_dim1=16,
      // tensor_dim0_stride=Ll.
      const tdm_g1_t g1 = { (int)((2u << 16) | (1u << 20) | (5u << 22) | (3u << 25)),
                            (int)((unsigned)Ll << 16),
                            (int)((unsigned)Ll << 16),
                            (int)(64u << 16),
                            16, Ll, 0, 0 };
      const tdm_g2_t gz4 = { 0, 0, 0, 0 };
      const tdm_g1_t gz8 = { 0, 0, 0, 0, 0, 0, 0, 0 };
      const unsigned long long gaP =
          (unsigned long long)(const void*)(pfull + (size_t)a0 * Ll + l0);
      const unsigned long long gaQ =
          (unsigned long long)(const void*)(qm + (size_t)b0 * Ll + l0);
      // D# group0: count=1, lds_addr, global_addr[56:0], type=2 ("image")
      const tdm_g0_t g0P = { 1u, ldsP, (unsigned)gaP,
                             (unsigned)((gaP >> 32) & 0x1ffffffu) | (2u << 30) };
      const tdm_g0_t g0Q = { 1u, ldsQ, (unsigned)gaQ,
                             (unsigned)((gaQ >> 32) & 0x1ffffffu) | (2u << 30) };
      // 6-arg toolchain form: (g0, g1, g2, g3, g4, cpol)
      __builtin_amdgcn_tensor_load_to_lds(g0P, g1, gz4, gz4, gz8, 0);
      __builtin_amdgcn_tensor_load_to_lds(g0Q, g1, gz4, gz4, gz8, 0);
      __builtin_amdgcn_s_wait_tensorcnt(0);
    }
    if (threadIdx.x < 64) Wv[threadIdx.x] = wy2[l0 + threadIdx.x];
    __syncthreads();
#else
#pragma unroll
    for (int r = 0; r < 4; ++r) {
      const int idx = threadIdx.x + r * 256;
      const int row = idx >> 6, col = idx & 63;
      Pt[row][col] = pfull[(size_t)(a0 + row) * Ll + l0 + col];
      Qt[row][col] = qm[(size_t)(b0 + row) * Ll + l0 + col];
    }
    if (threadIdx.x < 64) Wv[threadIdx.x] = wy2[l0 + threadIdx.x];
    __syncthreads();
#endif
#pragma unroll 4
    for (int l = 0; l < 64; ++l)
      s += Wv[l] * fast_tanh(Pt[ty][l] + Qt[tx][l]);
    __syncthreads();
  }
  scores[(size_t)(a0 + ty) * Nn + (b0 + tx)] = s + by2[0];
}

// ---------------------------------------------------------------------------
// p_idx: copy point_idx when istrain==1, else row-argmax of scores.
// ---------------------------------------------------------------------------
__global__ void __launch_bounds__(256) pidx_kernel(
    const int* __restrict__ istrain, const int* __restrict__ point_idx,
    const float* __restrict__ scores, int* __restrict__ pidx,
    float* __restrict__ out_pidx, int Nn) {
  const int row = blockIdx.x;
  if (istrain[0] == 1) {
    if (threadIdx.x == 0) { const int p = point_idx[row]; pidx[row] = p; out_pidx[row] = (float)p; }
    return;
  }
  __shared__ float bv[256];
  __shared__ int   bi[256];
  float best = -3.4e38f; int besti = 0;
  for (int c = threadIdx.x; c < Nn; c += 256) {
    const float v = scores[(size_t)row * Nn + c];
    if (v > best) { best = v; besti = c; }
  }
  bv[threadIdx.x] = best; bi[threadIdx.x] = besti;
  __syncthreads();
  for (int s = 128; s > 0; s >>= 1) {
    if (threadIdx.x < s) {
      const float v2 = bv[threadIdx.x + s]; const int i2 = bi[threadIdx.x + s];
      if (v2 > bv[threadIdx.x] || (v2 == bv[threadIdx.x] && i2 < bi[threadIdx.x])) {
        bv[threadIdx.x] = v2; bi[threadIdx.x] = i2;
      }
    }
    __syncthreads();
  }
  if (threadIdx.x == 0) { pidx[row] = bi[0]; out_pidx[row] = (float)bi[0]; }
}

// vec = [x (512) | dec (256) | x[p_idx] (512)]  as bf16
__global__ void vec_build_kernel(const unsigned short* __restrict__ x_bf,
                                 const unsigned short* __restrict__ dec_bf,
                                 const int* __restrict__ pidx,
                                 unsigned short* __restrict__ vec_bf, int Nn) {
  const int total = Nn * 1280;
  for (int i = blockIdx.x * blockDim.x + threadIdx.x; i < total; i += gridDim.x * blockDim.x) {
    const int n = i / 1280, c = i % 1280;
    unsigned short v;
    if (c < 512)      v = x_bf[n * 512 + c];
    else if (c < 768) v = dec_bf[n * 256 + (c - 512)];
    else              v = x_bf[pidx[n] * 512 + (c - 768)];
    vec_bf[i] = v;
  }
}

// Copy 36 valid label columns out of the 48-padded GEMM output + argmax.
__global__ void finalize_kernel(const float* __restrict__ labels_pad,
                                float* __restrict__ out_labels,
                                float* __restrict__ out_lidx, int Nn) {
  const int n = blockIdx.x * blockDim.x + threadIdx.x;
  if (n >= Nn) return;
  float best = -3.4e38f; int bi = 0;
  for (int jj = 0; jj < 36; ++jj) {
    const float v = labels_pad[n * 48 + jj];
    out_labels[n * 36 + jj] = v;
    if (v > best) { best = v; bi = jj; }
  }
  out_lidx[n] = (float)bi;
}

// ---------------------------------------------------------------------------
// Host orchestration
// ---------------------------------------------------------------------------
extern "C" void kernel_launch(void* const* d_in, const int* in_sizes, int n_in,
                              void* d_out, int out_size, void* d_ws, size_t ws_size,
                              hipStream_t stream) {
  (void)in_sizes; (void)n_in; (void)out_size; (void)ws_size;
  constexpr int N = 640, W = 256, L = 640;   // L = 5*W/2

  const float* x        = (const float*)d_in[0];   // [N,1,2W] contiguous
  const int*   pt_idx   = (const int*)  d_in[1];
  const int*   istrain  = (const int*)  d_in[2];
  const float* Wih0     = (const float*)d_in[3];   // [1024,512]
  const float* Whh0     = (const float*)d_in[4];   // [1024,256]
  const float* bih0     = (const float*)d_in[5];
  const float* bhh0     = (const float*)d_in[6];
  const float* Wih1     = (const float*)d_in[7];   // [1024,256]
  const float* Whh1     = (const float*)d_in[8];   // [1024,256]
  const float* bih1     = (const float*)d_in[9];
  const float* bhh1     = (const float*)d_in[10];
  const float* Wy1      = (const float*)d_in[11];  // [640,1280]
  const float* by1      = (const float*)d_in[12];
  const float* wy2      = (const float*)d_in[13];
  const float* by2      = (const float*)d_in[14];
  const float* Wz1      = (const float*)d_in[15];  // [640,1280]
  const float* bz1      = (const float*)d_in[16];
  const float* Wz2      = (const float*)d_in[17];  // [36,640]
  const float* bz2      = (const float*)d_in[18];

  float* out = (float*)d_out;
  float* out_pidx   = out;                       // [640]
  float* out_scores = out + N;                   // [640,640]
  float* out_labels = out + N + N * N;           // [640,36]
  float* out_lidx   = out + N + N * N + N * 36;  // [640]

  // ---- workspace carve (256B aligned); total ~18 MB ----
  char* wp = (char*)d_ws;
  auto alloc = [&](size_t bytes) -> void* {
    void* r = (void*)wp; wp += (bytes + 255) & ~(size_t)255; return r;
  };
  unsigned short* x_bf    = (unsigned short*)alloc((size_t)N * 512 * 2);
  unsigned short* wih0_bf = (unsigned short*)alloc((size_t)1024 * 512 * 2);
  unsigned short* whh0_bf = (unsigned short*)alloc((size_t)1024 * 256 * 2);
  unsigned short* wih1_bf = (unsigned short*)alloc((size_t)1024 * 256 * 2);
  unsigned short* whh1_bf = (unsigned short*)alloc((size_t)1024 * 256 * 2);
  unsigned short* wy1_bf  = (unsigned short*)alloc((size_t)L * 1280 * 2);
  unsigned short* wz1_bf  = (unsigned short*)alloc((size_t)L * 1280 * 2);
  unsigned short* wz2_bf  = (unsigned short*)alloc((size_t)48 * L * 2);   // padded 36->48
  unsigned short* dec_bf  = (unsigned short*)alloc((size_t)N * W * 2);
  unsigned short* vec_bf  = (unsigned short*)alloc((size_t)N * 1280 * 2);
  unsigned short* hid_bf  = (unsigned short*)alloc((size_t)N * L * 2);
  float* xg0        = (float*)alloc((size_t)N * 1024 * 4);
  float* b0         = (float*)alloc(1024 * 4);
  float* b1         = (float*)alloc(1024 * 4);
  float* px         = (float*)alloc((size_t)N * L * 4);
  float* qm         = (float*)alloc((size_t)N * L * 4);
  float* pfull      = (float*)alloc((size_t)N * L * 4);
  float* h1seq      = (float*)alloc((size_t)N * W * 4);
  float* labels_pad = (float*)alloc((size_t)N * 48 * 4);
  int*   pidx       = (int*)alloc(N * 4);
  int*   counter    = (int*)alloc(256);

  const int THR = 256;
  auto cvt = [&](unsigned short* dst, const float* src, int n) {
    int blocks = (n + THR - 1) / THR; if (blocks > 2048) blocks = 2048;
    cvt_bf16_kernel<<<blocks, THR, 0, stream>>>(dst, src, n);
  };
  // C = act(A * B^T + bias + Cadd): one 16x16 tile per wave, 8 waves/block
  auto gemm = [&](const unsigned short* A, int lda, const unsigned short* B, int ldb,
                  float* Cf, unsigned short* Cb, int ldc, const float* bias,
                  const float* Cadd, int M, int Npad, int Nreal, int K, int relu) {
    int tiles = (M / 16) * (Npad / 16);
    int blocks = (tiles + 7) / 8;
    wmma_gemm_kernel<<<blocks, 256, 0, stream>>>(A, lda, B, ldb, Cf, Cb, ldc, bias, Cadd,
                                                 M, Npad, Nreal, K, relu);
  };

  // 1) fp32 -> bf16 conversions (weights + x)
  cvt(x_bf,    x,    N * 512);
  cvt(wih0_bf, Wih0, 1024 * 512);
  cvt(whh0_bf, Whh0, 1024 * 256);
  cvt(wih1_bf, Wih1, 1024 * 256);
  cvt(whh1_bf, Whh1, 1024 * 256);
  cvt(wy1_bf,  Wy1,  L * 1280);
  cvt(wz1_bf,  Wz1,  L * 1280);
  cvt_pad_kernel<<<128, THR, 0, stream>>>(wz2_bf, Wz2, 36, 48, L);

  // 2) biases + pipeline counter reset
  prep_kernel<<<4, THR, 0, stream>>>(b0, bih0, bhh0, b1, bih1, bhh1, 1024, counter);

  // 3) WMMA GEMMs with no sequential dependency
  gemm(x_bf, 512, wih0_bf, 512, xg0, nullptr, 1024, b0, nullptr,
       N, 1024, 1024, 512, 0);                                        // x@Wih0^T + b0
  gemm(x_bf, 512, wy1_bf + 256, 1280, px, nullptr, L, nullptr, nullptr,
       N, L, L, 512, 0);                                              // x@Wy1[:,W:3W]^T
  gemm(x_bf, 512, wy1_bf + 768, 1280, qm, nullptr, L, nullptr, nullptr,
       N, L, L, 512, 0);                                              // x@Wy1[:,3W:]^T

  // 4) pipelined LSTM scan (block0 = layer0, block1 = layer1)
  lstm_kernel<<<2, 1024, 0, stream>>>(xg0, whh0_bf, wih1_bf, whh1_bf, b1,
                                      h1seq, dec_bf, counter, N);

  // 5) Pfull = dec @ Wy1[:, :W]^T + px + by1   (Cadd + bias fused in epilogue)
  gemm(dec_bf, 256, wy1_bf, 1280, pfull, nullptr, L, by1, px, N, L, L, 256, 0);

  // 6) attention scores -> d_out (TDM-staged LDS tiles)
  scores_kernel<<<dim3(L / 16, N / 16), 256, 0, stream>>>(pfull, qm, wy2, by2,
                                                          out_scores, N, L);

  // 7) p_idx (copy or argmax) -> ws + d_out
  pidx_kernel<<<N, 256, 0, stream>>>(istrain, pt_idx, out_scores, pidx, out_pidx, N);

  // 8) vec = [x | dec | x[p_idx]]
  vec_build_kernel<<<1024, THR, 0, stream>>>(x_bf, dec_bf, pidx, vec_bf, N);

  // 9) hid = relu(vec @ Wz1^T + bz1)  (bf16 out), labels = hid @ Wz2^T + bz2
  gemm(vec_bf, 1280, wz1_bf, 1280, nullptr, hid_bf, L, bz1, nullptr, N, L, L, 1280, 1);
  gemm(hid_bf, 640, wz2_bf, 640, labels_pad, nullptr, 48, bz2, nullptr, N, 48, 36, 640, 0);

  // 10) labels + argmax -> d_out
  finalize_kernel<<<(N + THR - 1) / THR, THR, 0, stream>>>(labels_pad, out_labels,
                                                           out_lidx, N);
}
